// ConceptBank_8753143349689
// MI455X (gfx1250) — matrix operand order repaired and checked by reading, not compiled, not run on hardware
//
#include <hip/hip_runtime.h>
#include <hip/hip_bf16.h>

// ---------------------------------------------------------------------------
// ConceptBank fused kernel for gfx1250 (MI455X, wave32, WMMA + TDM)
//   vec, p = softmax(l2norm(q) @ l2norm(emb)^T) -> (p @ emb, p)
// B=262144, N=512, D=256.
//
// Roofline: HBM floor ~1 GB (read q 256MB, write vec 256MB + p 512MB) ~44us.
// Design keys:
//  * bf16 WMMA 16x16x32, fp32 accum, for both matmuls (137 GFLOP total).
//  * 8-wave blocks (128 query rows). B-operands (bankn / blocked emb) staged
//    into LDS in 32KB slabs with the Tensor Data Mover (double buffered,
//    s_wait_tensorcnt): ~512KB of L2 operand traffic per 128 rows (~1 GB
//    total) instead of ~8 GB with per-wave streaming.
//  * sims are cosine similarities in [-1,1] => softmax with FIXED shift 1.0
//    is exact and stable: no row-max pass, no reductions in the WMMA loop.
//  * B fragments register-double-buffered so each WMMA has the next LDS
//    load pair already in flight (avoids s_wait_dscnt 0 serialization).
//  * p stored back into the LDS tile as bf16 => phase-2 A-fragments are two
//    raw ds_load_b128, no per-element converts.
// ---------------------------------------------------------------------------

typedef __attribute__((ext_vector_type(16))) __bf16       v16bf;
typedef __attribute__((ext_vector_type(8)))  float        v8f;
typedef __attribute__((ext_vector_type(4)))  unsigned int u32x4;
typedef __attribute__((ext_vector_type(8)))  int          i32x8;
typedef __attribute__((ext_vector_type(4)))  int          i32x4;

#define D_DIM 256
#define N_DIM 512
#define WAVES 8
#define ROWS_PER_WAVE 16
#define ROWS_PER_BLOCK (WAVES * ROWS_PER_WAVE)   // 128
#define SLAB_BYTES 32768
#define SLAB_HALFS (SLAB_BYTES / 2)              // 16384 bf16

union FragU {
    v16bf v;
    uint4 u[2];
};

// Load a 16-element bf16 fragment as two 16B vector loads (global or LDS).
__device__ __forceinline__ v16bf load_frag(const __bf16* p0, const __bf16* p1) {
    FragU f;
    f.u[0] = *(const uint4*)p0;
    f.u[1] = *(const uint4*)p1;
    return f.v;
}

// ---------------------------------------------------------------------------
// TDM: contiguous global -> LDS copy of nbytes (multiple of 8, <= 512KB).
// Descriptor per CDNA5 ISA (D# groups 0/1, 2D degenerate tile, 8B elements).
// Wave-level op: call from one wave only; EXEC ignored by hardware.
// This toolchain exposes the 6-arg builtin (g0, g1, g2, g3, g_extra, cpol).
// ---------------------------------------------------------------------------
__device__ __forceinline__ void tdm_copy_to_lds(const void* gsrc, unsigned int lds_off,
                                                unsigned int nbytes) {
    const unsigned long long ga = (unsigned long long)(uintptr_t)gsrc;
    const unsigned int ne = nbytes >> 3;           // 8-byte elements (fits 16b)
    u32x4 g0;
    g0[0] = 1u;                                    // count=1, user descriptor
    g0[1] = lds_off;                               // lds_addr
    g0[2] = (unsigned int)ga;                      // global_addr[31:0]
    g0[3] = (unsigned int)((ga >> 32) & 0x01FFFFFFu) | 0x80000000u;  // addr hi | type=2
    i32x8 g1;
    g1[0] = (int)(3u << 16);                       // data_size=3 (8B), no multicast
    g1[1] = (int)((ne & 0xFFFFu) << 16);           // tensor_dim0[15:0]
    g1[2] = (int)(((ne >> 16) & 0xFFFFu) | (1u << 16)); // dim0 hi | tensor_dim1=1 lo
    g1[3] = (int)((ne & 0xFFFFu) << 16);           // dim1 hi=0 | tile_dim0=ne
    g1[4] = 1;                                     // tile_dim1=1, tile_dim2=0
    g1[5] = (int)ne;                               // tensor_dim0_stride lo
    g1[6] = 0;                                     // stride hi / dim1_stride lo
    g1[7] = 0;
    const i32x4 z4 = {0, 0, 0, 0};
    const i32x8 z8 = {0, 0, 0, 0, 0, 0, 0, 0};
    __builtin_amdgcn_tensor_load_to_lds(g0, g1, z4, z4, z8, 0);
}

// ---------------------------------------------------------------------------
// Prep: bankn  = l2norm(emb) bf16, row-major (N x D)            [256 KB]
//       emb2   = emb blocked bf16 [kk][d][32], kk = n/32        [256 KB]
// one block (256 threads) per emb row
// ---------------------------------------------------------------------------
__global__ void cb_prep(const float* __restrict__ emb,
                        __bf16* __restrict__ bankn,
                        __bf16* __restrict__ emb2) {
    const int n = blockIdx.x;      // 0..511
    const int t = threadIdx.x;     // 0..255  (= d)
    float x = emb[n * D_DIM + t];
    float ss = x * x;
#pragma unroll
    for (int m = 16; m >= 1; m >>= 1) ss += __shfl_xor(ss, m, 32);
    __shared__ float wsum[8];
    if ((t & 31) == 0) wsum[t >> 5] = ss;
    __syncthreads();
    float tot = 0.f;
#pragma unroll
    for (int w = 0; w < 8; ++w) tot += wsum[w];
    const float inv = 1.0f / fmaxf(sqrtf(tot), 1e-12f);
    bankn[n * D_DIM + t] = (__bf16)(x * inv);
    emb2[(n >> 5) * (D_DIM * 32) + t * 32 + (n & 31)] = (__bf16)x;
}

// ---------------------------------------------------------------------------
// Main: 8 waves / block, 16 query rows per wave.
// ---------------------------------------------------------------------------
__global__ __launch_bounds__(256) void cb_main(const float* __restrict__ q,
                                               const __bf16* __restrict__ bankn,
                                               const __bf16* __restrict__ emb2,
                                               float* __restrict__ vout,
                                               float* __restrict__ pout) {
    __shared__ _Float16 simb[WAVES][ROWS_PER_WAVE][N_DIM];   // 128 KB sim/p tiles
    __shared__ __bf16   stage[2][SLAB_HALFS];                // 64 KB TDM slabs
    __shared__ float    ilb[WAVES][ROWS_PER_WAVE];           // 1/rowsum

    const int tid  = threadIdx.x;
    const int w    = tid >> 5;          // wave id 0..7
    const int lane = tid & 31;
    const int r    = lane & 15;         // row / col within 16x16 tile
    const int sel  = lane >> 4;         // lane half -> K sub-range
    const long qrow0 = (long)blockIdx.x * ROWS_PER_BLOCK + (long)w * ROWS_PER_WAVE;
    const bool issuer = (w == 0);

    // q staging aliases this wave's sim region (consumed before sim written;
    // same-wave DS ops are in order).
    __bf16* qs = (__bf16*)&simb[w][0][0];   // 16 x 256 bf16 = 8 KB

    // Prime: slab 0 of bankn in flight while we normalize q.
    if (issuer) tdm_copy_to_lds(bankn, (unsigned int)(uintptr_t)&stage[0][0], SLAB_BYTES);

    // ---- Phase 0: l2-normalize 16 query rows -> qs (bf16) -----------------
    {
        const float* qrow = q + (qrow0 + r) * D_DIM + sel * 128;
        float4 buf[32];
        float ss = 0.f;
#pragma unroll
        for (int j = 0; j < 32; ++j) {
            buf[j] = *(const float4*)(qrow + j * 4);
            ss += buf[j].x * buf[j].x + buf[j].y * buf[j].y +
                  buf[j].z * buf[j].z + buf[j].w * buf[j].w;
        }
        ss += __shfl_xor(ss, 16, 32);
        const float sc = 1.0f / fmaxf(sqrtf(ss), 1e-12f);
        __bf16* dst = qs + r * D_DIM + sel * 128;
#pragma unroll
        for (int j = 0; j < 32; ++j) {
            union { __bf16 h[4]; uint2 u; } pk;
            pk.h[0] = (__bf16)(buf[j].x * sc);
            pk.h[1] = (__bf16)(buf[j].y * sc);
            pk.h[2] = (__bf16)(buf[j].z * sc);
            pk.h[3] = (__bf16)(buf[j].w * sc);
            *(uint2*)(dst + j * 4) = pk.u;
        }
    }

    // A fragments: full K=256 of this wave's 16 q rows (64 VGPRs).
    // A k order used here (B uses the same): lane holds row r; e0..15 -> K=sel*16+e.
    v16bf a[8];
#pragma unroll
    for (int kc = 0; kc < 8; ++kc) {
        const __bf16* p0 = qs + r * D_DIM + kc * 32 + sel * 16;
        a[kc] = load_frag(p0, p0 + 8);
    }

    // ---- Phase 1: sim = qn @ bankn^T; per-lane partial exp-sums -----------
    // bankn consumed in 8 slabs of 64 rows (32 KB), TDM double-buffered.
    // softmax shift is the constant 1.0 (sims bounded by 1) => no max pass.
    float s8[8];
#pragma unroll
    for (int i = 0; i < 8; ++i) s8[i] = 0.f;

    for (int s = 0; s < 8; ++s) {
        const int buf = s & 1;
        if (issuer) {
            if (s < 7) {
                tdm_copy_to_lds(bankn + (s + 1) * 64 * D_DIM,
                                (unsigned int)(uintptr_t)&stage[buf ^ 1][0], SLAB_BYTES);
                __builtin_amdgcn_s_wait_tensorcnt(1);   // slab s landed
            } else {
                __builtin_amdgcn_s_wait_tensorcnt(0);
            }
        }
        __syncthreads();                                // slab s visible to all

        const __bf16* slab = &stage[buf][0];
        auto baddr = [&](int ncl, int kc) -> const __bf16* {
            return slab + (ncl * 16 + r) * D_DIM + kc * 32 + sel * 16;
        };
        // register double-buffer: one B pair always in flight
        v16bf bcur = load_frag(baddr(0, 0), baddr(0, 0) + 8);
        for (int ncl = 0; ncl < 4; ++ncl) {
            const int nc = s * 4 + ncl;                 // 16-col tile of N
            v8f c = {};
#pragma unroll
            for (int kc = 0; kc < 8; ++kc) {
                const __bf16* nx = (kc < 7) ? baddr(ncl, kc + 1)
                                 : (ncl < 3 ? baddr(ncl + 1, 0) : baddr(0, 0));
                v16bf bnxt = load_frag(nx, nx + 8);
                c = __builtin_amdgcn_wmma_f32_16x16x32_bf16(
                        false, a[kc], false, bcur, (short)0, c, false, false);
                bcur = bnxt;
            }
            // C layout: VGPR i -> row i + 8*sel, lane%16 -> col
#pragma unroll
            for (int i = 0; i < 8; ++i) {
                const float v = c[i];
                simb[w][i + sel * 8][nc * 16 + r] = (_Float16)v;
                s8[i] += __expf(v - 1.0f);
            }
        }
        __syncthreads();                                // done with slab s
    }

    // Reduce row sums across the 16 lanes of each half; publish 1/sum.
#pragma unroll
    for (int i = 0; i < 8; ++i) {
        float es = s8[i];
        es += __shfl_xor(es, 1, 32);
        es += __shfl_xor(es, 2, 32);
        es += __shfl_xor(es, 4, 32);
        es += __shfl_xor(es, 8, 32);
        s8[i] = es;
    }
    if (r == 0) {   // lanes 0 / 16 publish rows 0-7 / 8-15
#pragma unroll
        for (int i = 0; i < 8; ++i) ilb[w][i + sel * 8] = 1.0f / s8[i];
    }

    // Prime phase-2 slab 0 while p is finalized (stage unused here).
    if (issuer) tdm_copy_to_lds(emb2, (unsigned int)(uintptr_t)&stage[0][0], SLAB_BYTES);

    // ---- Phase 1.5: finalize p once: f32 to global, bf16 back to LDS ------
    for (int rr = 0; rr < ROWS_PER_WAVE; ++rr) {
        const float ir = ilb[w][rr];
        float* prow = pout + (qrow0 + rr) * (long)N_DIM;
#pragma unroll
        for (int it = 0; it < 2; ++it) {
            const int j = lane * 8 + it * 256;
            union { _Float16 h[8]; uint4 u; } sv;
            sv.u = *(const uint4*)&simb[w][rr][j];
            float p0 = __expf((float)sv.h[0] - 1.0f) * ir;
            float p1 = __expf((float)sv.h[1] - 1.0f) * ir;
            float p2 = __expf((float)sv.h[2] - 1.0f) * ir;
            float p3 = __expf((float)sv.h[3] - 1.0f) * ir;
            float p4 = __expf((float)sv.h[4] - 1.0f) * ir;
            float p5 = __expf((float)sv.h[5] - 1.0f) * ir;
            float p6 = __expf((float)sv.h[6] - 1.0f) * ir;
            float p7 = __expf((float)sv.h[7] - 1.0f) * ir;
            *(float4*)&prow[j]     = make_float4(p0, p1, p2, p3);
            *(float4*)&prow[j + 4] = make_float4(p4, p5, p6, p7);
            union { __bf16 h[8]; uint4 u; } pv;     // p as bf16 for the WMMA
            pv.h[0] = (__bf16)p0; pv.h[1] = (__bf16)p1;
            pv.h[2] = (__bf16)p2; pv.h[3] = (__bf16)p3;
            pv.h[4] = (__bf16)p4; pv.h[5] = (__bf16)p5;
            pv.h[6] = (__bf16)p6; pv.h[7] = (__bf16)p7;
            *(uint4*)&simb[w][rr][j] = pv.u;   // simb now holds p (bf16)
        }
    }

    // ---- Phase 2: vec = p @ emb ------------------------------------------
    // emb2 blocked [kk][256][32]; consumed in 8 slabs of 2 kk (32 KB each).
    v8f cvec[16];
#pragma unroll
    for (int dt = 0; dt < 16; ++dt) cvec[dt] = (v8f){0,0,0,0,0,0,0,0};

    for (int sp = 0; sp < 8; ++sp) {
        const int buf = sp & 1;
        if (issuer) {
            if (sp < 7) {
                tdm_copy_to_lds(emb2 + (sp + 1) * SLAB_HALFS,
                                (unsigned int)(uintptr_t)&stage[buf ^ 1][0], SLAB_BYTES);
                __builtin_amdgcn_s_wait_tensorcnt(1);
            } else {
                __builtin_amdgcn_s_wait_tensorcnt(0);
            }
        }
        __syncthreads();

        for (int kh = 0; kh < 2; ++kh) {
            const int kk = sp * 2 + kh;                 // K=32 slab over N
            // p A-fragment: same layout as q fragments, straight b128 loads
            const __bf16* pp = (const __bf16*)&simb[w][r][kk * 32] + sel * 16;
            v16bf pf = load_frag(pp, pp + 8);
            const __bf16* eb = &stage[buf][kh * (D_DIM * 32)];
            auto eaddr = [&](int dt) -> const __bf16* {
                return eb + (dt * 16 + r) * 32 + sel * 16;
            };
            v16bf bcur = load_frag(eaddr(0), eaddr(0) + 8);
#pragma unroll
            for (int dt = 0; dt < 16; ++dt) {
                const __bf16* nx = (dt < 15) ? eaddr(dt + 1) : eaddr(0);
                v16bf bnxt = load_frag(nx, nx + 8);
                cvec[dt] = __builtin_amdgcn_wmma_f32_16x16x32_bf16(
                               false, pf, false, bcur, (short)0, cvec[dt], false, false);
                bcur = bnxt;
            }
        }
        __syncthreads();
    }

    // ---- Phase 3: vec out via LDS transpose (coalesced float4 stores) ----
    // simb[w] (16 KB) is free now: reuse as 16 x 256 f32 staging.
    float* simf = (float*)&simb[w][0][0];
#pragma unroll
    for (int dt = 0; dt < 16; ++dt) {
#pragma unroll
        for (int i = 0; i < 8; ++i) {
            simf[(i + sel * 8) * D_DIM + dt * 16 + r] = cvec[dt][i];
        }
    }
    for (int rr = 0; rr < ROWS_PER_WAVE; ++rr) {
        float* vrow = vout + (qrow0 + rr) * (long)D_DIM;
#pragma unroll
        for (int it = 0; it < 2; ++it) {
            const int j = lane * 4 + it * 128;
            *(float4*)&vrow[j] = *(const float4*)&simf[rr * D_DIM + j];
        }
    }
}

// ---------------------------------------------------------------------------
extern "C" void kernel_launch(void* const* d_in, const int* in_sizes, int n_in,
                              void* d_out, int out_size, void* d_ws, size_t ws_size,
                              hipStream_t stream) {
    const float* q   = (const float*)d_in[0];   // (B, 256) fp32
    const float* emb = (const float*)d_in[1];   // (512, 256) fp32
    const int B = in_sizes[0] / D_DIM;          // 262144

    __bf16* bankn = (__bf16*)d_ws;                                        // 256 KB
    __bf16* emb2  = (__bf16*)((char*)d_ws + (size_t)N_DIM * D_DIM * 2);   // 256 KB

    float* vout = (float*)d_out;                       // (B, 256)
    float* pout = vout + (size_t)B * D_DIM;            // (B, 512)

    cb_prep<<<dim3(N_DIM), dim3(D_DIM), 0, stream>>>(emb, bankn, emb2);
    cb_main<<<dim3(B / ROWS_PER_BLOCK), dim3(256), 0, stream>>>(q, bankn, emb2, vout, pout);
}
